// Qwen3MoE_69715909149522
// MI455X (gfx1250) — compile-verified, matrix operand order checked by reading, not run on hardware
//
#include <hip/hip_runtime.h>
#include <hip/hip_bf16.h>
#include <stdint.h>

#define T_TOK 1024
#define H_DIM 2048
#define E_NUM 32
#define K_TOP 4
#define I_DIM 768
#define SI_DIM 4096
#define LSTR 36   // LDS row stride in halves: 32 data + 4 pad (72 B, 8B-aligned)

typedef __attribute__((ext_vector_type(16))) __bf16 v16bf;
typedef __attribute__((ext_vector_type(8)))  float  v8f;
typedef __attribute__((ext_vector_type(4)))  unsigned int v4u;
typedef __attribute__((ext_vector_type(8)))  int v8i;
typedef __attribute__((ext_vector_type(4)))  int v4i;

#define WMMA_BF16(a, b, c) \
  __builtin_amdgcn_wmma_f32_16x16x32_bf16(false, (a), false, (b), (short)0, (c), false, false)

__device__ __forceinline__ unsigned short f2bf(float x) {
  unsigned int u = __float_as_uint(x);
  u = (u + 0x7FFFu + ((u >> 16) & 1u)) >> 16;   // round-to-nearest-even bf16
  return (unsigned short)u;
}
__device__ __forceinline__ unsigned int pack2(float lo, float hi) {
  return (unsigned int)f2bf(lo) | ((unsigned int)f2bf(hi) << 16);
}
__device__ __forceinline__ unsigned long long pack4(float4 v) {
  return (unsigned long long)pack2(v.x, v.y) |
         ((unsigned long long)pack2(v.z, v.w) << 32);
}

union FragCvt { unsigned int u[8]; v16bf v; };

// 16x32 bf16 fragment from LDS laid out [row][k] (stride LSTR halves).
// ISA 7.12.2 16-bit A layout; B tiles staged transposed ([n][k]) reuse it.
__device__ __forceinline__ v16bf ld_frag(const unsigned short* lds, int tileRow,
                                         int lane) {
  const int m = lane & 15, hsel = lane >> 4;
  const unsigned short* p = lds + (size_t)(tileRow + m) * LSTR;
  FragCvt f;
#pragma unroll
  for (int v = 0; v < 8; ++v) {
    int k = ((v & 4) << 2) + hsel * 8 + ((v & 3) << 1);
    f.u[v] = *(const unsigned int*)(p + k);
  }
  return f.v;
}

// ---------------------------------------------------------------- utilities
__global__ void zero_counts_kernel(int* __restrict__ counts) {
  if (threadIdx.x < E_NUM) counts[threadIdx.x] = 0;
}

// ------------------------------------------------------------------- router
__global__ void router_topk_kernel(const float* __restrict__ x,
                                   const float* __restrict__ rw,
                                   int* __restrict__ counts,
                                   int* __restrict__ btok,
                                   float* __restrict__ bw) {
  const int t = blockIdx.x;
  const int e = threadIdx.x;                  // lane == expert (E == wave32)
  const float* xr = x + (size_t)t * H_DIM;
  float acc = 0.f;
#pragma unroll 4
  for (int h = 0; h < H_DIM; ++h)
    acc = __fmaf_rn(xr[h], rw[h * E_NUM + e], acc);

  float m = acc;
#pragma unroll
  for (int off = 16; off > 0; off >>= 1) m = fmaxf(m, __shfl_xor(m, off, 32));
  float p = __expf(acc - m);                  // softmax denom cancels in renorm

  int rank = 0;
#pragma unroll
  for (int j = 0; j < 32; ++j) {
    float pj = __shfl(p, j, 32);
    rank += (pj > p) || (pj == p && j < e);
  }
  const bool sel = rank < K_TOP;

  float ps = sel ? p : 0.f;
#pragma unroll
  for (int off = 16; off > 0; off >>= 1) ps += __shfl_xor(ps, off, 32);

  if (sel) {
    float w = p / ps;
    int pos = atomicAdd(&counts[e], 1);
    btok[e * T_TOK + pos] = t;
    bw[e * T_TOK + pos] = w;
  }
}

// -------------------------------------------------- shared expert: gate/up
// act[T,SI] = silu(x@ws_gate) * (x@ws_up).  128x64 tile, 8 waves (4x2).
__global__ void __launch_bounds__(256)
gemm_gateup_shared(const float* __restrict__ x, const float* __restrict__ wg,
                   const float* __restrict__ wu, float* __restrict__ act) {
  __shared__ __align__(16) unsigned short lA[128 * LSTR];
  __shared__ __align__(16) unsigned short lG[64 * LSTR];
  __shared__ __align__(16) unsigned short lU[64 * LSTR];
  const int tid = threadIdx.x, lane = tid & 31, wid = tid >> 5;
  const int wm = wid >> 1, wn = wid & 1;
  const int m0 = blockIdx.y * 128, n0 = blockIdx.x * 64;

  v8f ag[2][2] = {};
  v8f au[2][2] = {};
  for (int kb = 0; kb < H_DIM; kb += 32) {
    __syncthreads();
#pragma unroll
    for (int i = 0; i < 4; ++i) {           // A: 128x32 fp32, float4 loads
      int idx = tid + i * 256, r = idx >> 3, c4 = (idx & 7) * 4;
      float4 v = *(const float4*)(x + (size_t)(m0 + r) * H_DIM + kb + c4);
      *(unsigned long long*)(lA + r * LSTR + c4) = pack4(v);
    }
#pragma unroll
    for (int i = 0; i < 2; ++i) {           // B: 32x64 fp32 each, transposed
      int idx = tid + i * 256, kr = idx >> 4, nc4 = (idx & 15) * 4;
      float4 g = *(const float4*)(wg + (size_t)(kb + kr) * SI_DIM + n0 + nc4);
      float4 u = *(const float4*)(wu + (size_t)(kb + kr) * SI_DIM + n0 + nc4);
      lG[(nc4 + 0) * LSTR + kr] = f2bf(g.x);
      lG[(nc4 + 1) * LSTR + kr] = f2bf(g.y);
      lG[(nc4 + 2) * LSTR + kr] = f2bf(g.z);
      lG[(nc4 + 3) * LSTR + kr] = f2bf(g.w);
      lU[(nc4 + 0) * LSTR + kr] = f2bf(u.x);
      lU[(nc4 + 1) * LSTR + kr] = f2bf(u.y);
      lU[(nc4 + 2) * LSTR + kr] = f2bf(u.z);
      lU[(nc4 + 3) * LSTR + kr] = f2bf(u.w);
    }
    if (kb + 32 < H_DIM) {
      __builtin_prefetch(&wg[(size_t)(kb + 32 + (tid >> 6)) * SI_DIM + n0 + (tid & 63)], 0, 0);
      __builtin_prefetch(&wu[(size_t)(kb + 32 + (tid >> 6)) * SI_DIM + n0 + (tid & 63)], 0, 0);
    }
    __syncthreads();
    v16bf a0 = ld_frag(lA, wm * 32, lane);
    v16bf a1 = ld_frag(lA, wm * 32 + 16, lane);
#pragma unroll
    for (int tn = 0; tn < 2; ++tn) {
      v16bf bg = ld_frag(lG, wn * 32 + tn * 16, lane);
      v16bf bu = ld_frag(lU, wn * 32 + tn * 16, lane);
      ag[0][tn] = WMMA_BF16(a0, bg, ag[0][tn]);
      ag[1][tn] = WMMA_BF16(a1, bg, ag[1][tn]);
      au[0][tn] = WMMA_BF16(a0, bu, au[0][tn]);
      au[1][tn] = WMMA_BF16(a1, bu, au[1][tn]);
    }
  }
  const int hsel = lane >> 4, nl = lane & 15;
#pragma unroll
  for (int tm = 0; tm < 2; ++tm)
#pragma unroll
    for (int tn = 0; tn < 2; ++tn)
#pragma unroll
      for (int r = 0; r < 8; ++r) {
        int row = m0 + wm * 32 + tm * 16 + hsel * 8 + r;
        int col = n0 + wn * 32 + tn * 16 + nl;
        float g = ag[tm][tn][r], u = au[tm][tn][r];
        act[(size_t)row * SI_DIM + col] = g * u / (1.f + __expf(-g));
      }
}

// ----------------------------------------------------- shared expert: down
__global__ void __launch_bounds__(256)
gemm_down_shared(const float* __restrict__ act, const float* __restrict__ wd,
                 float* __restrict__ out) {
  __shared__ __align__(16) unsigned short lA[128 * LSTR];
  __shared__ __align__(16) unsigned short lB[128 * LSTR];
  const int tid = threadIdx.x, lane = tid & 31, wid = tid >> 5;
  const int wm = wid >> 1, wn = wid & 1;
  const int m0 = blockIdx.y * 128, n0 = blockIdx.x * 128;

  v8f acc[2][4] = {};
  for (int kb = 0; kb < SI_DIM; kb += 32) {
    __syncthreads();
#pragma unroll
    for (int i = 0; i < 4; ++i) {           // A: 128x32 fp32
      int idx = tid + i * 256, r = idx >> 3, c4 = (idx & 7) * 4;
      float4 v = *(const float4*)(act + (size_t)(m0 + r) * SI_DIM + kb + c4);
      *(unsigned long long*)(lA + r * LSTR + c4) = pack4(v);
    }
#pragma unroll
    for (int i = 0; i < 4; ++i) {           // B: 32x128 fp32, transposed
      int idx = tid + i * 256, kr = idx >> 5, nc4 = (idx & 31) * 4;
      float4 b = *(const float4*)(wd + (size_t)(kb + kr) * H_DIM + n0 + nc4);
      lB[(nc4 + 0) * LSTR + kr] = f2bf(b.x);
      lB[(nc4 + 1) * LSTR + kr] = f2bf(b.y);
      lB[(nc4 + 2) * LSTR + kr] = f2bf(b.z);
      lB[(nc4 + 3) * LSTR + kr] = f2bf(b.w);
    }
    if (kb + 32 < SI_DIM)
      __builtin_prefetch(&wd[(size_t)(kb + 32 + (tid >> 7)) * H_DIM + n0 + (tid & 127)], 0, 0);
    __syncthreads();
    v16bf a0 = ld_frag(lA, wm * 32, lane);
    v16bf a1 = ld_frag(lA, wm * 32 + 16, lane);
#pragma unroll
    for (int tn = 0; tn < 4; ++tn) {
      v16bf b = ld_frag(lB, wn * 64 + tn * 16, lane);
      acc[0][tn] = WMMA_BF16(a0, b, acc[0][tn]);
      acc[1][tn] = WMMA_BF16(a1, b, acc[1][tn]);
    }
  }
  const int hsel = lane >> 4, nl = lane & 15;
#pragma unroll
  for (int tm = 0; tm < 2; ++tm)
#pragma unroll
    for (int tn = 0; tn < 4; ++tn)
#pragma unroll
      for (int r = 0; r < 8; ++r) {
        int row = m0 + wm * 32 + tm * 16 + hsel * 8 + r;
        int col = n0 + wn * 64 + tn * 16 + nl;
        out[(size_t)row * H_DIM + col] = acc[tm][tn][r];
      }
}

// ------------------------------------------- routed experts: gate/up GEMM
__global__ void __launch_bounds__(256)
gemm_gateup_expert(const float* __restrict__ x, const float* __restrict__ wg_all,
                   const float* __restrict__ wu_all, const int* __restrict__ counts,
                   const int* __restrict__ btok, const float* __restrict__ bw,
                   unsigned short* __restrict__ hmid) {
  const int e = blockIdx.z;
  const int ne = counts[e];
  const int m0 = blockIdx.y * 128;
  if (m0 >= ne) return;
  const int n0 = blockIdx.x * 64;
  const float* wg = wg_all + (size_t)e * H_DIM * I_DIM;
  const float* wu = wu_all + (size_t)e * H_DIM * I_DIM;
  unsigned short* hm = hmid + (size_t)e * T_TOK * I_DIM;

  __shared__ __align__(16) unsigned short lA[128 * LSTR];
  __shared__ __align__(16) unsigned short lG[64 * LSTR];
  __shared__ __align__(16) unsigned short lU[64 * LSTR];
  __shared__ int   sTok[128];
  __shared__ float sW[128];
  const int tid = threadIdx.x, lane = tid & 31, wid = tid >> 5;
  const int wm = wid >> 1, wn = wid & 1;

  if (tid < 128) {
    int m = m0 + tid;
    bool ok = m < ne;
    sTok[tid] = ok ? btok[e * T_TOK + m] : 0;
    sW[tid]   = ok ? bw[e * T_TOK + m]   : 0.f;   // w=0 rows -> zero hmid
  }
  __syncthreads();

  v8f ag[2][2] = {};
  v8f au[2][2] = {};
  for (int kb = 0; kb < H_DIM; kb += 32) {
    __syncthreads();
#pragma unroll
    for (int i = 0; i < 4; ++i) {           // gathered A rows
      int idx = tid + i * 256, r = idx >> 3, c4 = (idx & 7) * 4;
      float4 v = *(const float4*)(x + (size_t)sTok[r] * H_DIM + kb + c4);
      *(unsigned long long*)(lA + r * LSTR + c4) = pack4(v);
    }
#pragma unroll
    for (int i = 0; i < 2; ++i) {
      int idx = tid + i * 256, kr = idx >> 4, nc4 = (idx & 15) * 4;
      float4 g = *(const float4*)(wg + (size_t)(kb + kr) * I_DIM + n0 + nc4);
      float4 u = *(const float4*)(wu + (size_t)(kb + kr) * I_DIM + n0 + nc4);
      lG[(nc4 + 0) * LSTR + kr] = f2bf(g.x);
      lG[(nc4 + 1) * LSTR + kr] = f2bf(g.y);
      lG[(nc4 + 2) * LSTR + kr] = f2bf(g.z);
      lG[(nc4 + 3) * LSTR + kr] = f2bf(g.w);
      lU[(nc4 + 0) * LSTR + kr] = f2bf(u.x);
      lU[(nc4 + 1) * LSTR + kr] = f2bf(u.y);
      lU[(nc4 + 2) * LSTR + kr] = f2bf(u.z);
      lU[(nc4 + 3) * LSTR + kr] = f2bf(u.w);
    }
    if (kb + 32 < H_DIM) {
      __builtin_prefetch(&wg[(size_t)(kb + 32 + (tid >> 6)) * I_DIM + n0 + (tid & 63)], 0, 0);
      __builtin_prefetch(&wu[(size_t)(kb + 32 + (tid >> 6)) * I_DIM + n0 + (tid & 63)], 0, 0);
    }
    __syncthreads();
    v16bf a0 = ld_frag(lA, wm * 32, lane);
    v16bf a1 = ld_frag(lA, wm * 32 + 16, lane);
#pragma unroll
    for (int tn = 0; tn < 2; ++tn) {
      v16bf bg = ld_frag(lG, wn * 32 + tn * 16, lane);
      v16bf bu = ld_frag(lU, wn * 32 + tn * 16, lane);
      ag[0][tn] = WMMA_BF16(a0, bg, ag[0][tn]);
      ag[1][tn] = WMMA_BF16(a1, bg, ag[1][tn]);
      au[0][tn] = WMMA_BF16(a0, bu, au[0][tn]);
      au[1][tn] = WMMA_BF16(a1, bu, au[1][tn]);
    }
  }
  const int hsel = lane >> 4, nl = lane & 15;
#pragma unroll
  for (int tm = 0; tm < 2; ++tm)
#pragma unroll
    for (int tn = 0; tn < 2; ++tn)
#pragma unroll
      for (int r = 0; r < 8; ++r) {
        int rl = wm * 32 + tm * 16 + hsel * 8 + r;
        int col = n0 + wn * 32 + tn * 16 + nl;
        float g = ag[tm][tn][r], u = au[tm][tn][r];
        float a = sW[rl] * g * u / (1.f + __expf(-g));
        hm[(size_t)(m0 + rl) * I_DIM + col] = f2bf(a);
      }
}

// ---------------------------------------------- routed experts: down GEMM
// A tile (bf16 hmid -> LDS) is moved by the Tensor Data Mover: 2D tile of
// 32 x 128 elements (data_size=2B), LDS padding 2 DW per 16 DW row so the
// landed stride is LSTR=36 halves, matching ld_frag.  Issued by wave 0,
// synchronized via s_wait_tensorcnt + workgroup barrier.
__global__ void __launch_bounds__(256)
gemm_down_expert(const unsigned short* __restrict__ hmid,
                 const float* __restrict__ wd_all, const int* __restrict__ counts,
                 const int* __restrict__ btok, float* __restrict__ out) {
  const int e = blockIdx.z;
  const int ne = counts[e];
  const int m0 = blockIdx.y * 128;
  if (m0 >= ne) return;
  const int n0 = blockIdx.x * 128;
  const float* wd = wd_all + (size_t)e * I_DIM * H_DIM;
  const unsigned short* hm = hmid + (size_t)e * T_TOK * I_DIM;

  __shared__ __align__(16) unsigned short lA[128 * LSTR];
  __shared__ __align__(16) unsigned short lB[128 * LSTR];
  __shared__ int sTok[128];
  const int tid = threadIdx.x, lane = tid & 31, wid = tid >> 5;
  const int wm = wid >> 1, wn = wid & 1;

  if (tid < 128) {
    int m = m0 + tid;
    sTok[tid] = (m < ne) ? btok[e * T_TOK + m] : 0;   // rows >= ne are zero
  }
  __syncthreads();

  const unsigned int ldsA_off = (unsigned int)(uintptr_t)&lA[0];

  v8f acc[2][4] = {};
  for (int kb = 0; kb < I_DIM; kb += 32) {
    __syncthreads();
    if (wid == 0) {
      // D# group0: count=1 | lds_addr | global_addr(57b) | type=2
      unsigned long long ga = (unsigned long long)(uintptr_t)hm +
                              ((unsigned long long)m0 * I_DIM + (unsigned)kb) * 2ull;
      v4u g0 = { 1u, ldsA_off, (unsigned int)(ga & 0xFFFFFFFFu),
                 ((unsigned int)(ga >> 32) & 0x01FFFFFFu) | 0x80000000u };
      // D# group1: data_size=2B(1), pad_enable, pad_interval=3(16DW),
      // pad_amount=1(2DW); tensor_dim0=768, tensor_dim1=1024,
      // tile_dim0=32, tile_dim1=128, dim0_stride=768
      v8i g1 = { (int)0x02D10000,
                 (int)(768u << 16),        // tensor_dim0[15:0]
                 (int)(1024u << 16),       // tensor_dim0[31:16]=0 | dim1[15:0]
                 (int)(32u << 16),         // dim1[31:16]=0 | tile_dim0=32
                 (int)128,                 // tile_dim1=128, tile_dim2=0
                 (int)768, 0, 0 };         // tensor_dim0_stride=768
      v4i gz = { 0, 0, 0, 0 };
      v8i gz8 = { 0, 0, 0, 0, 0, 0, 0, 0 };
      __builtin_amdgcn_tensor_load_to_lds(g0, g1, gz, gz, gz8, 0);
    }
#pragma unroll
    for (int i = 0; i < 4; ++i) {           // B: 32x128 fp32, transposed
      int idx = tid + i * 256, kr = idx >> 5, nc4 = (idx & 31) * 4;
      float4 b = *(const float4*)(wd + (size_t)(kb + kr) * H_DIM + n0 + nc4);
      lB[(nc4 + 0) * LSTR + kr] = f2bf(b.x);
      lB[(nc4 + 1) * LSTR + kr] = f2bf(b.y);
      lB[(nc4 + 2) * LSTR + kr] = f2bf(b.z);
      lB[(nc4 + 3) * LSTR + kr] = f2bf(b.w);
    }
    if (kb + 32 < I_DIM)
      __builtin_prefetch(&wd[(size_t)(kb + 32 + (tid >> 7)) * H_DIM + n0 + (tid & 127)], 0, 0);
    if (wid == 0) __builtin_amdgcn_s_wait_tensorcnt(0);
    __syncthreads();
    v16bf a0 = ld_frag(lA, wm * 32, lane);
    v16bf a1 = ld_frag(lA, wm * 32 + 16, lane);
#pragma unroll
    for (int tn = 0; tn < 4; ++tn) {
      v16bf b = ld_frag(lB, wn * 64 + tn * 16, lane);
      acc[0][tn] = WMMA_BF16(a0, b, acc[0][tn]);
      acc[1][tn] = WMMA_BF16(a1, b, acc[1][tn]);
    }
  }
  const int hsel = lane >> 4, nl = lane & 15;
#pragma unroll
  for (int tm = 0; tm < 2; ++tm)
#pragma unroll
    for (int tn = 0; tn < 4; ++tn)
#pragma unroll
      for (int r = 0; r < 8; ++r) {
        int rl = wm * 32 + tm * 16 + hsel * 8 + r;
        int col = n0 + wn * 64 + tn * 16 + nl;
        atomicAdd(&out[(size_t)sTok[rl] * H_DIM + col], acc[tm][tn][r]);
      }
}

// -------------------------------------------------------------------- host
extern "C" void kernel_launch(void* const* d_in, const int* in_sizes, int n_in,
                              void* d_out, int out_size, void* d_ws, size_t ws_size,
                              hipStream_t stream) {
  (void)in_sizes; (void)n_in; (void)out_size; (void)ws_size;
  const float* x       = (const float*)d_in[0];
  const float* rw      = (const float*)d_in[1];
  const float* w_gate  = (const float*)d_in[2];
  const float* w_up    = (const float*)d_in[3];
  const float* w_down  = (const float*)d_in[4];
  const float* ws_gate = (const float*)d_in[5];
  const float* ws_up   = (const float*)d_in[6];
  const float* ws_down = (const float*)d_in[7];
  float* out = (float*)d_out;

  char* ws = (char*)d_ws;
  int*   counts = (int*)ws;                                   // 128 B (pad 256)
  int*   btok   = (int*)(ws + 256);                           // E*T ints
  float* bw     = (float*)(ws + 256 + (size_t)E_NUM * T_TOK * 4);
  char*  p_act  = ws + 262400;                                // 256-aligned
  float* act    = (float*)p_act;                              // T*SI fp32 (16 MB)
  unsigned short* hmid =
      (unsigned short*)(p_act + (size_t)T_TOK * SI_DIM * 4);  // E*T*I bf16 (48 MB)

  zero_counts_kernel<<<1, 32, 0, stream>>>(counts);
  router_topk_kernel<<<T_TOK, 32, 0, stream>>>(x, rw, counts, btok, bw);

  gemm_gateup_shared<<<dim3(SI_DIM / 64, T_TOK / 128), 256, 0, stream>>>(
      x, ws_gate, ws_up, act);
  gemm_down_shared<<<dim3(H_DIM / 128, T_TOK / 128), 256, 0, stream>>>(
      act, ws_down, out);

  gemm_gateup_expert<<<dim3(I_DIM / 64, T_TOK / 128, E_NUM), 256, 0, stream>>>(
      x, w_gate, w_up, counts, btok, bw, hmid);
  gemm_down_expert<<<dim3(H_DIM / 128, T_TOK / 128, E_NUM), 256, 0, stream>>>(
      hmid, w_down, counts, btok, out);
}
